// AttnBlock_16037407883560
// MI455X (gfx1250) — matrix operand order repaired and checked by reading, not compiled, not run on hardware
//
#include <hip/hip_runtime.h>

typedef __bf16 bf16_t;
typedef bf16_t v16bf __attribute__((ext_vector_type(16)));
typedef bf16_t v8bf  __attribute__((ext_vector_type(8)));
typedef float  v8f   __attribute__((ext_vector_type(8)));

#define WMMA_BF16(a, b, c) \
  __builtin_amdgcn_wmma_f32_16x16x32_bf16(false, (a), false, (b), (short)0, (c), false, false)

static constexpr int Cdim = 512;
static constexpr int Ndim = 4096;   // 64*64 spatial
static constexpr int Bdim = 2;
static constexpr int NG   = 32;
static constexpr float SCALE = 0.04419417382415922f;  // 512^-0.5
static constexpr int SLD = 4100;    // padded LDS row stride (floats)

// round-to-nearest-even f32 -> bf16
__device__ __forceinline__ bf16_t f2bf(float f) {
  union { float f; unsigned u; } x; x.f = f;
  unsigned r = x.u + 0x7fffu + ((x.u >> 16) & 1u);
  unsigned short h = (unsigned short)(r >> 16);
  bf16_t o; __builtin_memcpy(&o, &h, 2);
  return o;
}

// A-operand fragment (16xK=32): row-major source, contraction contiguous.
// lane<16 holds K={0..7,16..23}, lane>=16 holds K={8..15,24..31} (ISA 7.12.2)
__device__ __forceinline__ v16bf ldfragA(const bf16_t* base, int ld, int lane) {
  const bf16_t* rp = base + (size_t)(lane & 15) * ld + ((lane >> 4) << 3);
  v8bf lo = *(const v8bf*)rp;
  v8bf hi = *(const v8bf*)(rp + 16);
  v16bf r;
#pragma unroll
  for (int i = 0; i < 8; ++i) { r[i] = lo[i]; r[i + 8] = hi[i]; }
  return r;
}

// B-operand fragment (K=32 x 16): source stored transposed row-major [n][k].
// lane<16 holds K={0..7,16..23}, lane>=16 holds K={8..15,24..31} (matches ldfragA)
__device__ __forceinline__ v16bf ldfragB(const bf16_t* base, int ld, int lane) {
  const bf16_t* rp = base + (size_t)(lane & 15) * ld + ((lane >> 4) << 3);
  v8bf lo = *(const v8bf*)rp;
  v8bf hi = *(const v8bf*)(rp + 16);
  v16bf r;
#pragma unroll
  for (int i = 0; i < 8; ++i) { r[i] = lo[i]; r[i + 8] = hi[i]; }
  return r;
}

__global__ __launch_bounds__(256) void k_f32_to_bf16(const float* __restrict__ src,
                                                     bf16_t* __restrict__ dst, int n) {
  int i = blockIdx.x * 256 + threadIdx.x;
  if (i < n) dst[i] = f2bf(src[i]);
}

// One WG per (b, g): mean / rstd over 16 channels x 4096 spatial (contiguous block).
__global__ __launch_bounds__(256) void k_gn_stats(const float* __restrict__ x,
                                                  float* __restrict__ stats) {
  __shared__ float ssum[256], ssq[256];
  const int bg = blockIdx.x;
  const float* p = x + (size_t)bg * (16 * Ndim);
  float s = 0.f, q = 0.f;
  for (int i = threadIdx.x; i < 16 * Ndim; i += 256) { float v = p[i]; s += v; q += v * v; }
  ssum[threadIdx.x] = s; ssq[threadIdx.x] = q;
  __syncthreads();
  for (int off = 128; off > 0; off >>= 1) {
    if (threadIdx.x < off) {
      ssum[threadIdx.x] += ssum[threadIdx.x + off];
      ssq[threadIdx.x]  += ssq[threadIdx.x + off];
    }
    __syncthreads();
  }
  if (threadIdx.x == 0) {
    float inv = 1.f / (16.f * Ndim);
    float mu  = ssum[0] * inv;
    float var = ssq[0] * inv - mu * mu;
    stats[bg * 2]     = mu;
    stats[bg * 2 + 1] = rsqrtf(var + 1e-6f);
  }
}

// Normalize + transpose: x[b][c][n] f32 -> hnT[b][n][c] bf16 (c contiguous for WMMA frags)
__global__ __launch_bounds__(256) void k_gn_apply_T(const float* __restrict__ x,
                                                    const float* __restrict__ stats,
                                                    const float* __restrict__ gamma,
                                                    const float* __restrict__ beta,
                                                    bf16_t* __restrict__ hnT) {
  __shared__ float tile[32][33];
  const int n0 = blockIdx.x * 32, c0 = blockIdx.y * 32, b = blockIdx.z;
  const int tx = threadIdx.x, ty = threadIdx.y;
#pragma unroll
  for (int i = 0; i < 4; ++i) {
    int c = c0 + ty + i * 8;
    tile[ty + i * 8][tx] = x[((size_t)b * Cdim + c) * Ndim + n0 + tx];
  }
  __syncthreads();
#pragma unroll
  for (int i = 0; i < 4; ++i) {
    int n = n0 + ty + i * 8;
    int c = c0 + tx;
    int g = b * NG + (c >> 4);
    float mu = stats[g * 2], rs = stats[g * 2 + 1];
    float v = (tile[tx][ty + i * 8] - mu) * rs * gamma[c] + beta[c];
    hnT[((size_t)b * Ndim + n) * Cdim + c] = f2bf(v);
  }
}

// D[M][Nn] = A[M][K] * B[Nn][K]^T  (+ optional row/col biases), bf16 in/out, f32 accum.
// Block = 8 waves; each wave owns a 32(M) x 64(Nn) tile (8 WMMA accumulators),
// with software-pipelined (double-buffered) fragment loads.
// Requires K % 64 == 0, K >= 64.
__global__ __launch_bounds__(256) void k_gemm_nt_bf16(
    const bf16_t* __restrict__ A, const bf16_t* __restrict__ B, bf16_t* __restrict__ out,
    int M, int Nn, int K,
    const float* __restrict__ bias_row, const float* __restrict__ bias_col,
    long long sA, long long sB, long long sO) {
  const int lane = threadIdx.x & 31, wave = threadIdx.x >> 5;
  const int b  = blockIdx.z;
  const int m0 = blockIdx.x * 64 + (wave >> 2) * 32;
  const int n0 = blockIdx.y * 256 + (wave & 3) * 64;
  const bf16_t* Ab = A + (size_t)b * sA + (size_t)m0 * K;
  const bf16_t* Bb = B + (size_t)b * sB + (size_t)n0 * K;
  bf16_t* Ob = out + (size_t)b * sO;
  v8f acc[8] = {};
  v16bf a00, a01, b00, b01, b02, b03;   // set 0
  v16bf a10, a11, b10, b11, b12, b13;   // set 1
#define LOADSET(s, kk)                                           \
  a##s##0 = ldfragA(Ab + (kk), K, lane);                         \
  a##s##1 = ldfragA(Ab + 16 * (size_t)K + (kk), K, lane);        \
  b##s##0 = ldfragB(Bb + (kk), K, lane);                         \
  b##s##1 = ldfragB(Bb + 16 * (size_t)K + (kk), K, lane);        \
  b##s##2 = ldfragB(Bb + 32 * (size_t)K + (kk), K, lane);        \
  b##s##3 = ldfragB(Bb + 48 * (size_t)K + (kk), K, lane)
#define WMMASET(s)                                               \
  acc[0] = WMMA_BF16(a##s##0, b##s##0, acc[0]);                  \
  acc[1] = WMMA_BF16(a##s##0, b##s##1, acc[1]);                  \
  acc[2] = WMMA_BF16(a##s##0, b##s##2, acc[2]);                  \
  acc[3] = WMMA_BF16(a##s##0, b##s##3, acc[3]);                  \
  acc[4] = WMMA_BF16(a##s##1, b##s##0, acc[4]);                  \
  acc[5] = WMMA_BF16(a##s##1, b##s##1, acc[5]);                  \
  acc[6] = WMMA_BF16(a##s##1, b##s##2, acc[6]);                  \
  acc[7] = WMMA_BF16(a##s##1, b##s##3, acc[7]);
  LOADSET(0, 0);
  int kk = 0;
  for (; kk + 64 < K; kk += 64) {
    LOADSET(1, kk + 32);   // prefetch next half-step
    WMMASET(0);
    LOADSET(0, kk + 64);   // prefetch next full-step
    WMMASET(1);
  }
  LOADSET(1, kk + 32);
  WMMASET(0);
  WMMASET(1);
#undef LOADSET
#undef WMMASET
  const int rb = (lane >> 4) * 8, cl = lane & 15;
#pragma unroll
  for (int half = 0; half < 2; ++half) {
#pragma unroll
    for (int t = 0; t < 4; ++t) {
      int n = n0 + t * 16 + cl;
      float bc = bias_col ? bias_col[n] : 0.f;
#pragma unroll
      for (int j = 0; j < 8; ++j) {
        int m = m0 + half * 16 + rb + j;
        float br = bias_row ? bias_row[m] : 0.f;
        Ob[(size_t)m * Nn + n] = f2bf(acc[half * 4 + t][j] + br + bc);
      }
    }
  }
}

// One WG per (b, 16-query tile). Full-row softmax with scores resident in 262KB LDS.
__global__ __launch_bounds__(256) void k_attention(
    const bf16_t* __restrict__ qT, const bf16_t* __restrict__ kT,
    const bf16_t* __restrict__ vm, bf16_t* __restrict__ hT) {
  extern __shared__ float S[];  // [16][SLD] f32 scores/probabilities
  const int lane = threadIdx.x & 31, wave = threadIdx.x >> 5;
  const int b  = blockIdx.y;
  const int q0 = blockIdx.x * 16;
  const size_t bNC = (size_t)b * Ndim * Cdim;

  // ---- scores: S[q][m] = (qT tile) . (kT rows), K = 512 ----
  {
    v16bf qf[16];
    const bf16_t* qb = qT + bNC + (size_t)q0 * Cdim;
#pragma unroll
    for (int ks = 0; ks < 16; ++ks) qf[ks] = ldfragA(qb + ks * 32, Cdim, lane);
    for (int kt = wave; kt < Ndim / 16; kt += 8) {
      v8f acc = {};
      const bf16_t* kb = kT + bNC + (size_t)kt * 16 * Cdim;
      v16bf kcur = ldfragB(kb, Cdim, lane);
#pragma unroll
      for (int ks = 0; ks < 16; ++ks) {
        v16bf knext;
        if (ks < 15) knext = ldfragB(kb + (ks + 1) * 32, Cdim, lane);
        acc = WMMA_BF16(qf[ks], kcur, acc);
        if (ks < 15) kcur = knext;
      }
      int col = kt * 16 + (lane & 15);
      int rbs = (lane >> 4) * 8;
#pragma unroll
      for (int j = 0; j < 8; ++j) S[(rbs + j) * SLD + col] = acc[j] * SCALE;
    }
  }
  __syncthreads();

  // ---- softmax over keys: each wave owns 2 query rows ----
  for (int r = wave * 2; r < wave * 2 + 2; ++r) {
    float* row = S + r * SLD;
    float m = -3.0e38f;
    for (int i = lane; i < Ndim; i += 32) m = fmaxf(m, row[i]);
#pragma unroll
    for (int off = 16; off > 0; off >>= 1) m = fmaxf(m, __shfl_xor(m, off, 32));
    float s = 0.f;
    for (int i = lane; i < Ndim; i += 32) { float e = __expf(row[i] - m); row[i] = e; s += e; }
#pragma unroll
    for (int off = 16; off > 0; off >>= 1) s += __shfl_xor(s, off, 32);
    float inv = 1.f / s;
    for (int i = lane; i < Ndim; i += 32) row[i] *= inv;
  }
  __syncthreads();

  // ---- attend: O[q][c] = sum_m P[q][m] * v[c][m]; wave owns 64 c columns ----
  const int c0 = wave * 64;
  const bf16_t* vb = vm + (size_t)b * Cdim * Ndim;
  v8f o[4] = {};
  v16bf p0, v00, v01, v02, v03;   // set 0
  v16bf p1, v10, v11, v12, v13;   // set 1
#define PFRAG(dst, mt)                                                          \
  {                                                                             \
    const float* sp = S + (lane & 15) * SLD + (mt) + ((lane >> 4) << 3);        \
    _Pragma("unroll")                                                           \
    for (int i = 0; i < 8; ++i) { dst[i] = f2bf(sp[i]); dst[i + 8] = f2bf(sp[i + 16]); } \
  }
#define VSET(s, mt)                                                             \
  v##s##0 = ldfragB(vb + (size_t)(c0 +  0) * Ndim + (mt), Ndim, lane);          \
  v##s##1 = ldfragB(vb + (size_t)(c0 + 16) * Ndim + (mt), Ndim, lane);          \
  v##s##2 = ldfragB(vb + (size_t)(c0 + 32) * Ndim + (mt), Ndim, lane);          \
  v##s##3 = ldfragB(vb + (size_t)(c0 + 48) * Ndim + (mt), Ndim, lane)
#define OSET(s)                                                                 \
  o[0] = WMMA_BF16(p##s, v##s##0, o[0]);                                        \
  o[1] = WMMA_BF16(p##s, v##s##1, o[1]);                                        \
  o[2] = WMMA_BF16(p##s, v##s##2, o[2]);                                        \
  o[3] = WMMA_BF16(p##s, v##s##3, o[3]);
  PFRAG(p0, 0); VSET(0, 0);
  int mt = 0;
  for (; mt + 64 < Ndim; mt += 64) {
    PFRAG(p1, mt + 32); VSET(1, mt + 32);
    OSET(0);
    PFRAG(p0, mt + 64); VSET(0, mt + 64);
    OSET(1);
  }
  PFRAG(p1, mt + 32); VSET(1, mt + 32);
  OSET(0);
  OSET(1);
#undef PFRAG
#undef VSET
#undef OSET
  bf16_t* hb = hT + bNC + (size_t)q0 * Cdim;
  const int rbs = (lane >> 4) * 8, cl = lane & 15;
#pragma unroll
  for (int t = 0; t < 4; ++t)
#pragma unroll
    for (int j = 0; j < 8; ++j)
      hb[(size_t)(rbs + j) * Cdim + c0 + t * 16 + cl] = f2bf(o[t][j]);
}

// out[b][d][n] = x + wo @ h + bo : D[n][d] tiles, 32x64 per wave, fused residual epilogue.
__global__ __launch_bounds__(256) void k_outproj(
    const bf16_t* __restrict__ hT, const bf16_t* __restrict__ wo,
    const float* __restrict__ bo, const float* __restrict__ x,
    float* __restrict__ out) {
  const int lane = threadIdx.x & 31, wave = threadIdx.x >> 5;
  const int b  = blockIdx.z;
  const int n0 = blockIdx.x * 64 + (wave >> 2) * 32;
  const int d0 = blockIdx.y * 256 + (wave & 3) * 64;
  const bf16_t* Ab = hT + (size_t)b * Ndim * Cdim + (size_t)n0 * Cdim;
  const bf16_t* Bb = wo + (size_t)d0 * Cdim;
  const int K = Cdim;
  v8f acc[8] = {};
  v16bf a00, a01, b00, b01, b02, b03;
  v16bf a10, a11, b10, b11, b12, b13;
#define LOADSET(s, kk)                                           \
  a##s##0 = ldfragA(Ab + (kk), K, lane);                         \
  a##s##1 = ldfragA(Ab + 16 * (size_t)K + (kk), K, lane);        \
  b##s##0 = ldfragB(Bb + (kk), K, lane);                         \
  b##s##1 = ldfragB(Bb + 16 * (size_t)K + (kk), K, lane);        \
  b##s##2 = ldfragB(Bb + 32 * (size_t)K + (kk), K, lane);        \
  b##s##3 = ldfragB(Bb + 48 * (size_t)K + (kk), K, lane)
#define WMMASET(s)                                               \
  acc[0] = WMMA_BF16(a##s##0, b##s##0, acc[0]);                  \
  acc[1] = WMMA_BF16(a##s##0, b##s##1, acc[1]);                  \
  acc[2] = WMMA_BF16(a##s##0, b##s##2, acc[2]);                  \
  acc[3] = WMMA_BF16(a##s##0, b##s##3, acc[3]);                  \
  acc[4] = WMMA_BF16(a##s##1, b##s##0, acc[4]);                  \
  acc[5] = WMMA_BF16(a##s##1, b##s##1, acc[5]);                  \
  acc[6] = WMMA_BF16(a##s##1, b##s##2, acc[6]);                  \
  acc[7] = WMMA_BF16(a##s##1, b##s##3, acc[7]);
  LOADSET(0, 0);
  int kk = 0;
  for (; kk + 64 < K; kk += 64) {
    LOADSET(1, kk + 32);
    WMMASET(0);
    LOADSET(0, kk + 64);
    WMMASET(1);
  }
  LOADSET(1, kk + 32);
  WMMASET(0);
  WMMASET(1);
#undef LOADSET
#undef WMMASET
  const int rb = (lane >> 4) * 8, cl = lane & 15;
#pragma unroll
  for (int half = 0; half < 2; ++half) {
#pragma unroll
    for (int t = 0; t < 4; ++t) {
      int d = d0 + t * 16 + cl;
      float bias = bo[d];
      size_t base = ((size_t)b * Cdim + d) * Ndim + n0 + half * 16 + rb;
      v8f a = acc[half * 4 + t];
      float4 x0 = *(const float4*)(x + base);
      float4 x1 = *(const float4*)(x + base + 4);
      float4 r0 = make_float4(a[0] + bias + x0.x, a[1] + bias + x0.y,
                              a[2] + bias + x0.z, a[3] + bias + x0.w);
      float4 r1 = make_float4(a[4] + bias + x1.x, a[5] + bias + x1.y,
                              a[6] + bias + x1.z, a[7] + bias + x1.w);
      *(float4*)(out + base)     = r0;
      *(float4*)(out + base + 4) = r1;
    }
  }
}

extern "C" void kernel_launch(void* const* d_in, const int* in_sizes, int n_in,
                              void* d_out, int out_size, void* d_ws, size_t ws_size,
                              hipStream_t stream) {
  const float* x   = (const float*)d_in[0];
  const float* gns = (const float*)d_in[1];
  const float* gnb = (const float*)d_in[2];
  const float* wq  = (const float*)d_in[3];
  const float* bq  = (const float*)d_in[4];
  const float* wk  = (const float*)d_in[5];
  const float* bk  = (const float*)d_in[6];
  const float* wv  = (const float*)d_in[7];
  const float* bv  = (const float*)d_in[8];
  const float* wo  = (const float*)d_in[9];
  const float* bo  = (const float*)d_in[10];
  float* out = (float*)d_out;

  const size_t WW = (size_t)Cdim * Cdim;   // 262144
  const size_t NC = (size_t)Ndim * Cdim;   // 2097152
  bf16_t* ws  = (bf16_t*)d_ws;
  bf16_t* wqb = ws;
  bf16_t* wkb = wqb + WW;
  bf16_t* wvb = wkb + WW;
  bf16_t* wob = wvb + WW;
  bf16_t* hnT = wob + WW;                  // [B][N][C]
  bf16_t* qT  = hnT + (size_t)Bdim * NC;   // [B][N][C]
  bf16_t* kT  = qT  + (size_t)Bdim * NC;   // [B][N][C]
  bf16_t* vm  = kT  + (size_t)Bdim * NC;   // [B][C][N]
  bf16_t* hT  = vm  + (size_t)Bdim * NC;   // [B][N][C]
  float* stats = (float*)(hT + (size_t)Bdim * NC);  // [B*G][mu, rstd]

  k_f32_to_bf16<<<(int)(WW / 256), 256, 0, stream>>>(wq, wqb, (int)WW);
  k_f32_to_bf16<<<(int)(WW / 256), 256, 0, stream>>>(wk, wkb, (int)WW);
  k_f32_to_bf16<<<(int)(WW / 256), 256, 0, stream>>>(wv, wvb, (int)WW);
  k_f32_to_bf16<<<(int)(WW / 256), 256, 0, stream>>>(wo, wob, (int)WW);

  k_gn_stats<<<Bdim * NG, 256, 0, stream>>>(x, stats);
  k_gn_apply_T<<<dim3(Ndim / 32, Cdim / 32, Bdim), dim3(32, 8), 0, stream>>>(x, stats, gns, gnb, hnT);

  // q, k: [N][C] = hnT[N][C] x w[C_out][C]^T   (bias over columns)
  k_gemm_nt_bf16<<<dim3(Ndim / 64, Cdim / 256, Bdim), 256, 0, stream>>>(
      hnT, wqb, qT, Ndim, Cdim, Cdim, nullptr, bq, (long long)NC, 0LL, (long long)NC);
  k_gemm_nt_bf16<<<dim3(Ndim / 64, Cdim / 256, Bdim), 256, 0, stream>>>(
      hnT, wkb, kT, Ndim, Cdim, Cdim, nullptr, bk, (long long)NC, 0LL, (long long)NC);
  // v: [C][N] = wv[C][C] x hnT[N][C]^T        (bias over rows)
  k_gemm_nt_bf16<<<dim3(Cdim / 64, Ndim / 256, Bdim), 256, 0, stream>>>(
      wvb, hnT, vm, Cdim, Ndim, Cdim, bv, nullptr, 0LL, (long long)NC, (long long)NC);

  k_attention<<<dim3(Ndim / 16, Bdim), 256, 16 * SLD * sizeof(float), stream>>>(qT, kT, vm, hT);

  k_outproj<<<dim3(Ndim / 64, Cdim / 256, Bdim), 256, 0, stream>>>(hT, wob, bo, x, out);
}
